// HybridPooling_40346922778975
// MI455X (gfx1250) — matrix operand order repaired and checked by reading, not compiled, not run on hardware
//
#include <hip/hip_runtime.h>
#include <hip/hip_bf16.h>
#include <math.h>

#define CIN 256
#define HID 256
#define GSEG 4096
#define LDSW 264   // padded column stride (bf16 elems): 132 dwords -> conflict-free strided LDS reads

typedef __attribute__((ext_vector_type(16))) __bf16 v16bf;
typedef __attribute__((ext_vector_type(8)))  float  v8f;
typedef __attribute__((ext_vector_type(4)))  int    v4i;

typedef __attribute__((address_space(1))) const char g_char;
typedef __attribute__((address_space(3))) char       l_char;
typedef __attribute__((address_space(1))) v4i        g_v4i;
typedef __attribute__((address_space(3))) v4i        l_v4i;

#if defined(__gfx1250__) && __has_builtin(__builtin_amdgcn_global_load_async_to_lds_b128)
#define HAVE_ASYNC_LDS 1
#else
#define HAVE_ASYNC_LDS 0
#endif

__device__ __forceinline__ v8f wmma_bf16(const v16bf& a, const v16bf& b, const v8f& c) {
    return __builtin_amdgcn_wmma_f32_16x16x32_bf16(false, a, false, b, (short)0, c, false, false);
}

// ---------------------------------------------------------------- kernel 0a: clear accumulators
__global__ void zero_ws_kernel(unsigned* __restrict__ p, long long n) {
    long long i = (long long)blockIdx.x * blockDim.x + threadIdx.x;
    long long stride = (long long)gridDim.x * blockDim.x;
    for (; i < n; i += stride) p[i] = 0u;
}

// ---------------------------------------------------------------- kernel 0b: pre-convert weights
// Column-major bf16 copies so B-fragments are contiguous 16-byte loads (and the attention
// kernel's LDS staging becomes a raw async DMA with no conversion).
__global__ void convert_weights_kernel(const float* __restrict__ W1a, const float* __restrict__ W1f,
                                       const float* __restrict__ W2f, __bf16* __restrict__ W1abf,
                                       __bf16* __restrict__ W1fbf, __bf16* __restrict__ W2fbf)
{
    const int i = blockIdx.x * blockDim.x + threadIdx.x;
    const int stride = gridDim.x * blockDim.x;
    for (int j = i; j < HID * LDSW; j += stride) {           // W1a [256k x 256n] -> [n][264]
        const int n = j / LDSW, k = j % LDSW;
        W1abf[j] = (k < CIN) ? (__bf16)W1a[k * HID + n] : (__bf16)0.0f;
    }
    for (int j = i; j < 768 * HID; j += stride) {            // W1f [768k x 256n] -> [n][768]
        const int n = j / 768, k = j % 768;
        W1fbf[j] = (__bf16)W1f[k * HID + n];
    }
    for (int j = i; j < HID * CIN; j += stride) {            // W2f [256k x 256n] -> [n][256]
        const int n = j >> 8, k = j & 255;
        W2fbf[j] = (__bf16)W2f[k * CIN + n];
    }
}

// ---------------------------------------------------------------- pooling flush helper
__device__ __forceinline__ void flush_seg(int g, int lane, float runLen,
                                          const float* aS, const float* aA, const float* aM,
                                          float* __restrict__ att, float* __restrict__ sums,
                                          unsigned* __restrict__ maxenc, float* __restrict__ counts)
{
#pragma unroll
    for (int j = 0; j < 8; ++j) {
        const int c = j * 32 + lane;
        atomicAdd(&sums[(size_t)g * CIN + c], aS[j]);
        atomicAdd(&att [(size_t)g * CIN + c], aA[j]);
        const unsigned bits = __float_as_uint(aM[j]);
        const unsigned enc  = (bits & 0x80000000u) ? ~bits : (bits | 0x80000000u);
        atomicMax(&maxenc[(size_t)g * CIN + c], enc);
    }
    if (lane == 0) atomicAdd(&counts[g], runLen);
}

// ---------------------------------------------------------------- kernel 1
// Fused: h = relu(x@W1a + b1a); a = sigmoid(h@W2a + b2a); run-aggregated segment sum/att/max/count.
__global__ __launch_bounds__(128) void attn_pool_kernel(
    const float* __restrict__ x, const int* __restrict__ batch,
    const __bf16* __restrict__ W1abf, const float* __restrict__ b1a,
    const float* __restrict__ W2a, const float* __restrict__ b2a,
    float* __restrict__ att, float* __restrict__ sums,
    unsigned* __restrict__ maxenc, float* __restrict__ counts, int nNodes)
{
    extern __shared__ __bf16 Wlds[];   // [n][LDSW] column-major bf16, 132 KB

#if HAVE_ASYNC_LDS
    {   // Raw async DMA of the pre-converted weight image into LDS (ASYNCcnt-tracked).
        g_char* gsrc = (g_char*)(unsigned long long)(const char*)W1abf;
        l_char* ldst = (l_char*)(unsigned int)(unsigned long long)(char*)&Wlds[0];
        const int nchunks = (HID * LDSW * 2) / 16;           // 8448 x b128
        for (int j = threadIdx.x; j < nchunks; j += 128)
            __builtin_amdgcn_global_load_async_to_lds_b128((g_v4i*)(gsrc + j * 16),
                                                           (l_v4i*)(ldst + j * 16), 0, 0);
#if __has_builtin(__builtin_amdgcn_s_wait_asynccnt)
        __builtin_amdgcn_s_wait_asynccnt(0);
#else
        asm volatile("s_wait_asynccnt 0x0" ::: "memory");
#endif
    }
#else
    for (int i = threadIdx.x; i < HID * LDSW; i += 128) Wlds[i] = W1abf[i];
#endif
    __syncthreads();

    const int wave   = threadIdx.x >> 5;
    const int lane   = threadIdx.x & 31;
    const int lane15 = lane & 15;
    const int hi     = lane >> 4;

    const int row0 = (blockIdx.x * 4 + wave) * 16;
    if (row0 >= nNodes) return;   // no further barriers below

    const int myrow = row0 + lane15;
    const int ldrow = (myrow < nNodes) ? myrow : (nNodes - 1);
    const float* xr = x + (size_t)ldrow * CIN;

    // 16 accumulators (full H strip), kc-outer / t-inner, double-buffered B fragments.
    v8f acc[16];
#pragma unroll
    for (int t = 0; t < 16; ++t) acc[t] = v8f{0.f,0.f,0.f,0.f,0.f,0.f,0.f,0.f};

#pragma unroll
    for (int kc = 0; kc < 8; ++kc) {
        const int kb  = kc * 32 + (hi ? 16 : 0);
        const int klo = kc * 32 + (hi ? 16 : 0);
        v16bf a;
#pragma unroll
        for (int e = 0; e < 8; ++e) {
            a[e]     = (__bf16)xr[kb + e];
            a[e + 8] = (__bf16)xr[kb + 8 + e];
        }
        v16bf bbuf[2];
        {   const __bf16* wcol = &Wlds[lane15 * LDSW + klo];
#pragma unroll
            for (int e = 0; e < 16; ++e) bbuf[0][e] = wcol[e];
        }
#pragma unroll
        for (int t = 0; t < 16; ++t) {
            if (t < 15) {   // prefetch next B fragment while current WMMA executes
                const __bf16* wcol = &Wlds[((t + 1) * 16 + lane15) * LDSW + klo];
#pragma unroll
                for (int e = 0; e < 16; ++e) bbuf[(t + 1) & 1][e] = wcol[e];
            }
            acc[t] = wmma_bf16(a, bbuf[t & 1], acc[t]);
        }
    }

    // bias + relu, fold into h@W2a partial dot, cross-lane reduce, sigmoid.
    float partial[8];
#pragma unroll
    for (int v = 0; v < 8; ++v) partial[v] = 0.0f;
#pragma unroll
    for (int t = 0; t < 16; ++t) {
        const int n = t * 16 + lane15;
        const float bias = b1a[n];
        const float w2   = W2a[n];
#pragma unroll
        for (int v = 0; v < 8; ++v) {
            float hv = acc[t][v] + bias;
            hv = hv > 0.0f ? hv : 0.0f;
            partial[v] += hv * w2;
        }
    }
    float sig[8];
    const float b2 = b2a[0];
#pragma unroll
    for (int v = 0; v < 8; ++v) {
        float p = partial[v];
        p += __shfl_xor(p, 1, 32);
        p += __shfl_xor(p, 2, 32);
        p += __shfl_xor(p, 4, 32);
        p += __shfl_xor(p, 8, 32);
        sig[v] = 1.0f / (1.0f + __expf(-(p + b2)));
    }

    // Segment accumulation: batch is sorted, so aggregate runs of equal g in registers
    // and flush once per run (~16x fewer global atomics).
    float aS[8], aA[8], aM[8];
    int gcur = -1;
    float runLen = 0.0f;
#pragma unroll
    for (int m = 0; m < 16; ++m) {
        const int row = row0 + m;
        if (row >= nNodes) break;
        const float am = __shfl(sig[m & 7], (m < 8) ? 0 : 16, 32);
        const int g = batch[row];
        if (g != gcur) {
            if (runLen > 0.0f) flush_seg(gcur, lane, runLen, aS, aA, aM, att, sums, maxenc, counts);
            gcur = g; runLen = 0.0f;
#pragma unroll
            for (int j = 0; j < 8; ++j) { aS[j] = 0.0f; aA[j] = 0.0f; aM[j] = -INFINITY; }
        }
        const float* xrow = x + (size_t)row * CIN;
#pragma unroll
        for (int j = 0; j < 8; ++j) {
            const float xv = xrow[j * 32 + lane];
            aS[j] += xv;
            aA[j] += am * xv;
            aM[j]  = fmaxf(aM[j], xv);
        }
        runLen += 1.0f;
    }
    if (runLen > 0.0f) flush_seg(gcur, lane, runLen, aS, aA, aM, att, sums, maxenc, counts);
}

// ---------------------------------------------------------------- kernel 2
__global__ void combine_kernel(const float* __restrict__ att, const float* __restrict__ sums,
                               const unsigned* __restrict__ maxenc, const float* __restrict__ counts,
                               __bf16* __restrict__ comb)
{
    const int idx = blockIdx.x * blockDim.x + threadIdx.x;
    if (idx >= GSEG * CIN) return;
    const int g = idx >> 8;
    const int c = idx & 255;
    const float cnt  = counts[g];
    const float a    = att[idx];
    const float mean = sums[idx] / fmaxf(cnt, 1.0f);
    float mx = 0.0f;
    if (cnt > 0.0f) {
        const unsigned enc  = maxenc[idx];
        const unsigned bits = (enc & 0x80000000u) ? (enc & 0x7FFFFFFFu) : ~enc;
        mx = __uint_as_float(bits);
    }
    __bf16* row = comb + (size_t)g * (3 * CIN);
    row[c]           = (__bf16)a;
    row[CIN + c]     = (__bf16)mean;
    row[2 * CIN + c] = (__bf16)mx;
}

// ---------------------------------------------------------------- kernel 3
// out = relu(combined @ W1f + b1f) @ W2f + b2f  (one wave per 16 graphs, bf16 weights)
__global__ __launch_bounds__(32) void final_mlp_kernel(
    const __bf16* __restrict__ comb, const __bf16* __restrict__ W1fbf,
    const float* __restrict__ b1f, const __bf16* __restrict__ W2fbf,
    const float* __restrict__ b2f, float* __restrict__ out)
{
    __shared__ __bf16 Hlds[16 * LDSW];
    const int lane   = threadIdx.x & 31;
    const int lane15 = lane & 15;
    const int hi     = lane >> 4;
    const int g0     = blockIdx.x * 16;

    const __bf16* arow = comb + (size_t)(g0 + lane15) * (3 * CIN);

    // Stage 1: 768 -> 256, relu, transpose into A-friendly LDS layout.
#pragma unroll 1
    for (int t = 0; t < 16; ++t) {
        const int n = t * 16 + lane15;
        const __bf16* bcol = W1fbf + (size_t)n * 768;
        v8f acc = {0.f,0.f,0.f,0.f,0.f,0.f,0.f,0.f};
#pragma unroll
        for (int kc = 0; kc < 24; ++kc) {
            const int kb  = kc * 32 + (hi ? 16 : 0);
            const int klo = kc * 32 + (hi ? 16 : 0);
            v16bf a, b;
#pragma unroll
            for (int e = 0; e < 8; ++e) {
                a[e]     = arow[kb + e];
                a[e + 8] = arow[kb + 8 + e];
            }
#pragma unroll
            for (int e = 0; e < 16; ++e) b[e] = bcol[klo + e];
            acc = wmma_bf16(a, b, acc);
        }
        const float bias = b1f[n];
#pragma unroll
        for (int v = 0; v < 8; ++v) {
            float hv = acc[v] + bias;
            Hlds[(v + hi * 8) * LDSW + n] = (__bf16)(hv > 0.0f ? hv : 0.0f);
        }
    }
    __syncthreads();

    // Stage 2: 256 -> 256, add bias, store fp32.
    const __bf16* hrow = &Hlds[lane15 * LDSW];
#pragma unroll 1
    for (int t = 0; t < 16; ++t) {
        const int n = t * 16 + lane15;
        const __bf16* bcol = W2fbf + (size_t)n * 256;
        v8f acc = {0.f,0.f,0.f,0.f,0.f,0.f,0.f,0.f};
#pragma unroll
        for (int kc = 0; kc < 8; ++kc) {
            const int kb  = kc * 32 + (hi ? 16 : 0);
            const int klo = kc * 32 + (hi ? 16 : 0);
            v16bf a, b;
#pragma unroll
            for (int e = 0; e < 8; ++e) {
                a[e]     = hrow[kb + e];
                a[e + 8] = hrow[kb + 8 + e];
            }
#pragma unroll
            for (int e = 0; e < 16; ++e) b[e] = bcol[klo + e];
            acc = wmma_bf16(a, b, acc);
        }
        const float bias = b2f[n];
#pragma unroll
        for (int v = 0; v < 8; ++v)
            out[(size_t)(g0 + v + hi * 8) * CIN + n] = acc[v] + bias;
    }
}

// ---------------------------------------------------------------- launcher
extern "C" void kernel_launch(void* const* d_in, const int* in_sizes, int n_in,
                              void* d_out, int out_size, void* d_ws, size_t ws_size,
                              hipStream_t stream)
{
    const float* x     = (const float*)d_in[0];
    const int*   batch = (const int*)d_in[1];
    const float* W1a   = (const float*)d_in[2];
    const float* b1a   = (const float*)d_in[3];
    const float* W2a   = (const float*)d_in[4];
    const float* b2a   = (const float*)d_in[5];
    const float* W1f   = (const float*)d_in[6];
    const float* b1f   = (const float*)d_in[7];
    const float* W2f   = (const float*)d_in[8];
    const float* b2f   = (const float*)d_in[9];
    const int nNodes = in_sizes[0] / CIN;   // 500000

    // Workspace layout: att[G*C]f | sums[G*C]f | maxenc[G*C]u | counts[G]f |
    //                   comb[G*768]bf16 | W1abf[256*264]bf16 | W1fbf[256*768]bf16 | W2fbf[256*256]bf16
    char* ws = (char*)d_ws;
    float*    att    = (float*)ws;
    float*    sums   = att + (size_t)GSEG * CIN;
    unsigned* maxenc = (unsigned*)(sums + (size_t)GSEG * CIN);
    float*    counts = (float*)(maxenc + (size_t)GSEG * CIN);
    __bf16*   comb   = (__bf16*)(counts + GSEG);
    __bf16*   W1abf  = comb  + (size_t)GSEG * 3 * CIN;
    __bf16*   W1fbf  = W1abf + (size_t)HID * LDSW;
    __bf16*   W2fbf  = W1fbf + (size_t)HID * 768;

    const long long nzero = 3LL * GSEG * CIN + GSEG;     // att..counts are contiguous dwords
    zero_ws_kernel<<<1024, 256, 0, stream>>>((unsigned*)ws, nzero);
    convert_weights_kernel<<<256, 256, 0, stream>>>(W1a, W1f, W2f, W1abf, W1fbf, W2fbf);

    const int tiles = (nNodes + 63) / 64;                // 64 nodes per 128-thread block
    const size_t w1aLds = (size_t)HID * LDSW * sizeof(__bf16);   // 132 KB dynamic LDS
    attn_pool_kernel<<<tiles, 128, w1aLds, stream>>>(x, batch, W1abf, b1a, W2a, b2a,
                                                     att, sums, maxenc, counts, nNodes);

    combine_kernel<<<(GSEG * CIN + 255) / 256, 256, 0, stream>>>(att, sums, maxenc, counts, comb);

    final_mlp_kernel<<<GSEG / 16, 32, 0, stream>>>(comb, W1fbf, b1f, W2fbf, b2f, (float*)d_out);
}
